// RelationOp_73693048864902
// MI455X (gfx1250) — matrix operand order repaired and checked by reading, not compiled, not run on hardware
//
#include <hip/hip_runtime.h>
#include <hip/hip_bf16.h>
#include <math.h>

#define N_NODES 50000
#define N_EDGES 1600000
#define HIDDEN 128
#define HEADS 8
#define PER_HEAD 16
#define NEG_SLOPE 0.2f
#define MT_PER_BLOCK 5            // 50000 / 16 = 3125 = 625 * 5 (remainder-free)

typedef __attribute__((ext_vector_type(16))) __bf16 v16bf;
typedef __attribute__((ext_vector_type(8)))  float  v8f;

// Monotonic order-preserving f32 <-> u32 mapping (for atomicMax on floats incl. negatives)
__device__ __forceinline__ unsigned enc_f32(float x) {
    int i = __float_as_int(x);
    return (i < 0) ? ~(unsigned)i : ((unsigned)i | 0x80000000u);
}
__device__ __forceinline__ float dec_f32(unsigned u) {
    int i = (u & 0x80000000u) ? (int)(u & 0x7fffffffu) : ~(int)u;
    return __int_as_float(i);
}

// ---------------------------------------------------------------- init scratch
__global__ void init_ws(float* __restrict__ agg, float* __restrict__ denom,
                        unsigned* __restrict__ emax_u) {
    int i = blockIdx.x * blockDim.x + threadIdx.x;
    if (i < N_NODES * HIDDEN) agg[i] = 0.0f;
    if (i < N_NODES * HEADS) {
        denom[i]  = 0.0f;
        emax_u[i] = enc_f32(-INFINITY);
    }
}

// ---------------------------------------------------------------- z = feat @ W (bf16 WMMA, f32 acc)
// 625 blocks x 256 threads (8 waves). Wave w owns N-tile w (== head w).
// W is staged into LDS pre-swizzled into B-fragment layout [ntile][kc][lane][16],
// so each wave's 4 B fragments are two contiguous ds_load_b128 per fragment, then
// live in VGPRs across all 5 M-tiles handled by the block.
__global__ __launch_bounds__(256) void gemm_wmma(const float* __restrict__ feat,
                                                 const float* __restrict__ W,
                                                 float* __restrict__ z) {
    __shared__ __bf16 Wswz[HEADS * 4 * 32 * 16];     // 32 KB of 320 KB WGP LDS
    // cooperative load + convert + swizzle into fragment layout
    for (int i = threadIdx.x; i < HEADS * 4 * 32 * 16; i += 256) {
        int e     = i & 15;
        int ln    = (i >> 4) & 31;
        int kc    = (i >> 9) & 3;
        int nt    = i >> 11;
        int koffB = (ln >= 16) ? 16 : 0;             // B: contiguous K = 16*(lane/16) + e, matches A
        int k     = kc * 32 + koffB + e;
        int col   = nt * 16 + (ln & 15);
        Wswz[i] = (__bf16)W[k * HIDDEN + col];
    }
    __syncthreads();

    const int lane = threadIdx.x & 31;
    const int wave = threadIdx.x >> 5;               // 0..7 == N tile == head
    const int col  = wave * 16 + (lane & 15);

    // hoist the 4 B fragments into VGPRs (32 B contiguous per lane per fragment)
    const v16bf b0 = *(const v16bf*)&Wswz[((wave * 4 + 0) * 32 + lane) * 16];
    const v16bf b1 = *(const v16bf*)&Wswz[((wave * 4 + 1) * 32 + lane) * 16];
    const v16bf b2 = *(const v16bf*)&Wswz[((wave * 4 + 2) * 32 + lane) * 16];
    const v16bf b3 = *(const v16bf*)&Wswz[((wave * 4 + 3) * 32 + lane) * 16];

    const int koffA = (lane >= 16) ? 16 : 0;         // 16-bit A layout (contiguous K, matches B)

#pragma unroll
    for (int mt = 0; mt < MT_PER_BLOCK; ++mt) {
        const int mbase = (blockIdx.x * MT_PER_BLOCK + mt) * 16;
        const float* arow = feat + (size_t)(mbase + (lane & 15)) * HIDDEN;
        if (mt + 1 < MT_PER_BLOCK)                   // global_prefetch_b8 of next A tile
            __builtin_prefetch(arow + 16 * HIDDEN, 0, 0);

        v8f acc = {};
#pragma unroll
        for (int kc = 0; kc < 4; ++kc) {             // fully unrolled -> 4 static WMMAs
            v16bf a;
#pragma unroll
            for (int e = 0; e < 8; ++e)              // VGPR 0..3: K = koffA + 0..7
                a[e] = (__bf16)arow[kc * 32 + koffA + e];
#pragma unroll
            for (int e = 8; e < 16; ++e)             // VGPR 4..7: K = koffA + 8..15
                a[e] = (__bf16)arow[kc * 32 + koffA + e];
            const v16bf bf = (kc == 0) ? b0 : (kc == 1) ? b1 : (kc == 2) ? b2 : b3;
            acc = __builtin_amdgcn_wmma_f32_16x16x32_bf16(
                false, a, false, bf, (short)0, acc, false, false);
        }
        // D layout: VGPR r -> M = r (lanes 0-15) / r+8 (lanes 16-31), N = col
        const int mo = mbase + ((lane >= 16) ? 8 : 0);
#pragma unroll
        for (int r = 0; r < 8; ++r)
            z[(size_t)(mo + r) * HIDDEN + col] = acc[r];
    }
}

// ---------------------------------------------------------------- el/er per (node, head)
__global__ void node_logits(const float* __restrict__ z,
                            const float* __restrict__ attn_l,
                            const float* __restrict__ attn_r,
                            float* __restrict__ el, float* __restrict__ er) {
    int i = blockIdx.x * blockDim.x + threadIdx.x;     // i = n*HEADS + h
    if (i >= N_NODES * HEADS) return;
    int h = i & (HEADS - 1);
    const float* zp = z + (size_t)i * PER_HEAD;        // n*128 + h*16 == i*16
    const float* al = attn_l + h * PER_HEAD;
    const float* ar = attn_r + h * PER_HEAD;
    float sl = 0.0f, sr = 0.0f;
#pragma unroll
    for (int p = 0; p < PER_HEAD; ++p) {
        float v = zp[p];
        sl += v * al[p];
        sr += v * ar[p];
    }
    el[i] = sl;
    er[i] = sr;
}

// ---------------------------------------------------------------- segment max over dst (atomic)
__global__ void edge_max(const int* __restrict__ src, const int* __restrict__ dst,
                         const float* __restrict__ el, const float* __restrict__ er,
                         unsigned* __restrict__ emax_u) {
    int e = blockIdx.x * blockDim.x + threadIdx.x;
    if (e >= N_EDGES) return;
    int s = src[e], d = dst[e];
#pragma unroll
    for (int h = 0; h < HEADS; ++h) {
        float x = el[s * HEADS + h] + er[d * HEADS + h];
        x = (x > 0.0f) ? x : NEG_SLOPE * x;            // leaky relu
        atomicMax(emax_u + d * HEADS + h, enc_f32(x));
    }
}

// ---------------------------------------------------------------- decode max, empty segment -> 0
__global__ void decode_max(const unsigned* __restrict__ emax_u, float* __restrict__ emax) {
    int i = blockIdx.x * blockDim.x + threadIdx.x;
    if (i >= N_NODES * HEADS) return;
    float m = dec_f32(emax_u[i]);
    emax[i] = isfinite(m) ? m : 0.0f;
}

// ---------------------------------------------------------------- exp, denom + weighted aggregation
// one thread per (edge, head): recompute score (L2-resident reads), 1+16 f32 atomics.
__global__ void edge_agg(const int* __restrict__ src, const int* __restrict__ dst,
                         const float* __restrict__ el, const float* __restrict__ er,
                         const float* __restrict__ emax, const float* __restrict__ z,
                         float* __restrict__ denom, float* __restrict__ agg) {
    int t = blockIdx.x * blockDim.x + threadIdx.x;     // t = e*HEADS + h
    if (t >= N_EDGES * HEADS) return;
    int e = t >> 3, h = t & (HEADS - 1);
    int s = src[e], d = dst[e];
    float x = el[s * HEADS + h] + er[d * HEADS + h];
    x = (x > 0.0f) ? x : NEG_SLOPE * x;
    float a = __expf(x - emax[d * HEADS + h]);
    atomicAdd(denom + d * HEADS + h, a);
    const float4* zp = (const float4*)(z + (size_t)s * HIDDEN + h * PER_HEAD);
    float* ap = agg + (size_t)d * HIDDEN + h * PER_HEAD;
#pragma unroll
    for (int q = 0; q < 4; ++q) {
        float4 v = zp[q];
        atomicAdd(ap + 4 * q + 0, v.x * a);
        atomicAdd(ap + 4 * q + 1, v.y * a);
        atomicAdd(ap + 4 * q + 2, v.z * a);
        atomicAdd(ap + 4 * q + 3, v.w * a);
    }
}

// ---------------------------------------------------------------- normalize + bias
__global__ void finalize(const float* __restrict__ agg, const float* __restrict__ denom,
                         const float* __restrict__ bias, float* __restrict__ out) {
    int i = blockIdx.x * blockDim.x + threadIdx.x;
    if (i >= N_NODES * HIDDEN) return;
    int f = i & (HIDDEN - 1);
    int n = i >> 7;
    int h = f >> 4;
    float dn = fmaxf(denom[n * HEADS + h], 1e-9f);
    out[i] = agg[i] / dn + bias[f];
}

extern "C" void kernel_launch(void* const* d_in, const int* in_sizes, int n_in,
                              void* d_out, int out_size, void* d_ws, size_t ws_size,
                              hipStream_t stream) {
    const float* feat   = (const float*)d_in[0];
    const int*   src    = (const int*)  d_in[1];
    const int*   dst    = (const int*)  d_in[2];
    const float* W      = (const float*)d_in[3];
    const float* attn_l = (const float*)d_in[4];
    const float* attn_r = (const float*)d_in[5];
    const float* bias   = (const float*)d_in[6];
    float* out = (float*)d_out;

    // workspace carve-up (~59.2 MB total)
    float*    z      = (float*)d_ws;                          // 50000*128
    float*    agg    = z   + (size_t)N_NODES * HIDDEN;        // 50000*128
    float*    el     = agg + (size_t)N_NODES * HIDDEN;        // 50000*8
    float*    er     = el  + (size_t)N_NODES * HEADS;         // 50000*8
    unsigned* emax_u = (unsigned*)(er + (size_t)N_NODES * HEADS);
    float*    emax   = (float*)(emax_u + (size_t)N_NODES * HEADS);
    float*    denom  = emax + (size_t)N_NODES * HEADS;

    const int B = 256;
    init_ws    <<<(N_NODES * HIDDEN + B - 1) / B, B, 0, stream>>>(agg, denom, emax_u);
    gemm_wmma  <<<(N_NODES / 16) / MT_PER_BLOCK, 256, 0, stream>>>(feat, W, z);
    node_logits<<<(N_NODES * HEADS + B - 1) / B, B, 0, stream>>>(z, attn_l, attn_r, el, er);
    edge_max   <<<(N_EDGES + B - 1) / B, B, 0, stream>>>(src, dst, el, er, emax_u);
    decode_max <<<(N_NODES * HEADS + B - 1) / B, B, 0, stream>>>(emax_u, emax);
    edge_agg   <<<(N_EDGES * HEADS + B - 1) / B, B, 0, stream>>>(src, dst, el, er, emax, z, denom, agg);
    finalize   <<<(N_NODES * HIDDEN + B - 1) / B, B, 0, stream>>>(agg, denom, bias, out);
}
